// SGFormer_14594298872383
// MI455X (gfx1250) — matrix operand order repaired and checked by reading, not compiled, not run on hardware
//
#include <hip/hip_runtime.h>
#include <hip/hip_bf16.h>
#include <stdint.h>

typedef __attribute__((ext_vector_type(16))) _Float16 v16h;
typedef __attribute__((ext_vector_type(2)))  _Float16 v2h;
typedef __attribute__((ext_vector_type(8)))  float    v8f;

constexpr int   kN       = 100000;
constexpr int   kInF     = 512;
constexpr int   kHid     = 256;
constexpr int   kClasses = 40;
constexpr float kEps     = 1e-5f;
constexpr float kNf      = 100000.0f;

// ---------------------------------------------------------------------------
// WMMA 16x16x32 f16 -> f32
// ---------------------------------------------------------------------------
static __device__ __forceinline__ v8f wmma_f16(v16h a, v16h b, v8f c) {
  return __builtin_amdgcn_wmma_f32_16x16x32_f16(false, a, false, b, (short)0, c,
                                                false, false);
}

// Fragment-order index maps (inverse of the CDNA5 16-bit WMMA VGPR layouts).
// A (16x32, M x K): lane = ((k>>3)&1)*16 + m ; elem e = (k>>4)*8 + (k&7)
static __device__ __forceinline__ int a_frag_idx(int m, int k) {
  int lane = ((((k >> 3) & 1) << 4) + m);
  int e = (((k >> 4) << 3) + (k & 7));
  return lane * 16 + e;
}
// B (32x16, K x N): lane = ((k>>3)&1)*16 + n ; elem e = (k>>4)*8 + (k&7)
static __device__ __forceinline__ int b_frag_idx(int n, int k) {
  int lane = ((((k >> 3) & 1) << 4) + n);
  int e = (((k >> 4) << 3) + (k & 7));
  return lane * 16 + e;
}
// A-panel global layout for f16 activations: [row>>4][col>>5][512] fragments.
static __device__ __forceinline__ size_t a_panel_idx(size_t row, int col, int kts) {
  return ((row >> 4) * kts + (col >> 5)) * 512 + a_frag_idx((int)(row & 15), col & 31);
}

// Row-wise LayerNorm over 256 cols staged in LDS; wave w handles rows 4w..4w+3.
// Writes f32 (residual path) and f16 A-panel-swizzled (GEMM operand) copies.
static __device__ __forceinline__ void ln_store32(const float* sOut, float* dst,
                                                  _Float16* dstH, size_t rowBase,
                                                  const float* g, const float* b,
                                                  int wave, int lane, bool relu) {
#pragma unroll
  for (int rr = 0; rr < 4; ++rr) {
    const int m = wave * 4 + rr;
    float vals[8];
    float s = 0.f, s2 = 0.f;
#pragma unroll
    for (int j = 0; j < 8; ++j) {
      float v = sOut[m * 256 + lane + 32 * j];
      vals[j] = v; s += v; s2 += v * v;
    }
#pragma unroll
    for (int o = 16; o > 0; o >>= 1) {
      s  += __shfl_xor(s, o, 32);
      s2 += __shfl_xor(s2, o, 32);
    }
    const float mu  = s * (1.f / 256.f);
    const float var = s2 * (1.f / 256.f) - mu * mu;
    const float rs  = rsqrtf(var + kEps);
#pragma unroll
    for (int j = 0; j < 8; ++j) {
      const int c = lane + 32 * j;
      float o = g[c] * (vals[j] - mu) * rs + b[c];
      if (relu) o = fmaxf(o, 0.f);
      dst[(rowBase + m) * 256 + c] = o;
      dstH[a_panel_idx(rowBase + m, c, 8)] = (_Float16)o;
    }
  }
}

// ---------------------------------------------------------------------------
// Workhorse GEMM: block = 256 thr (8 waves), 32 rows x 256 cols, K tiles of 32.
// Per wave: 2 row-tiles x 2 col-tiles = 4 WMMAs per K-tile.
// AF16: A is panel-swizzled f16 in global -> barrier-free direct loads.
// !AF16 (input layer): A is f32 row-major, staged via LDS swizzle.
// B pre-swizzled in global: [KD/32][16 col-tiles][512].
// ---------------------------------------------------------------------------
enum { EPI_LN_RELU = 0, EPI_Q = 1, EPI_K = 2, EPI_V = 3, EPI_ATTN = 4 };

template <int EPI, bool AF16, int KD>
__global__ __launch_bounds__(256) void gemm_rows32(
    const void* __restrict__ Aptr,        // A operand (layout per AF16)
    const _Float16* __restrict__ W,       // B swizzled: [KD/32][16][512]
    const float* __restrict__ bias,       // [256] (QKV / input layer)
    float* __restrict__ outF,             // f32 out (LN variants)
    _Float16* __restrict__ outH,          // f16 out (K/V row-major; Q/h panel)
    const float* __restrict__ g,          // LN gamma
    const float* __restrict__ bta,        // LN beta
    const _Float16* __restrict__ Vb,      // V (f16 row-major) for attention
    const float* __restrict__ prev,       // residual input for attention (f32)
    const float* __restrict__ denomB,     // per-row normalizer (EPI_ATTN)
    float* __restrict__ statSum,          // atomic scalar (sum of squares)
    float* __restrict__ colSum)           // atomic [256] column sums (K only)
{
  __shared__ __align__(32) _Float16 sA[2 * 512];  // used only when !AF16
  __shared__ float sOut[32 * 256];
  __shared__ float sRed[256];
  __shared__ float sRow[32];

  const int tid  = threadIdx.x;
  const int lane = tid & 31;
  const int wave = tid >> 5;
  const size_t rowBase = (size_t)blockIdx.x * 32;

  v8f acc00 = {}, acc01 = {}, acc10 = {}, acc11 = {};

  if (AF16) {
    // ---- barrier-free main loop: direct global fragment loads ----
    const v16h* Ap = (const v16h*)Aptr;
    const v16h* Wf = (const v16h*)W;
    const size_t p0 = rowBase >> 4;                 // panel of rows 0..15
    constexpr int KT = KD / 32;
    for (int kt = 0; kt < KT; ++kt) {
      v16h a0 = Ap[(p0 * KT + kt) * 32 + lane];
      v16h a1 = Ap[((p0 + 1) * KT + kt) * 32 + lane];
      v16h b0 = Wf[(size_t)(kt * 16 + 2 * wave) * 32 + lane];
      v16h b1 = Wf[(size_t)(kt * 16 + 2 * wave + 1) * 32 + lane];
      acc00 = wmma_f16(a0, b0, acc00);
      acc01 = wmma_f16(a0, b1, acc01);
      acc10 = wmma_f16(a1, b0, acc10);
      acc11 = wmma_f16(a1, b1, acc11);
    }
  } else {
    // ---- f32 A staged through LDS in fragment order ----
    int aDst[4];
#pragma unroll
    for (int j = 0; j < 4; ++j) {
      int r = wave + 8 * j;
      aDst[j] = (r >> 4) * 512 + a_frag_idx(r & 15, lane);
    }
    const float* A = (const float*)Aptr;
    const v16h* Wf = (const v16h*)W;
    for (int kt = 0; kt < KD / 32; ++kt) {
      const int k0 = kt * 32;
#pragma unroll
      for (int j = 0; j < 4; ++j)
        sA[aDst[j]] = (_Float16)A[(rowBase + wave + 8 * j) * KD + k0 + lane];
      __syncthreads();
      v16h a0 = ((const v16h*)sA)[lane];
      v16h a1 = ((const v16h*)sA)[32 + lane];
      v16h b0 = Wf[(size_t)(kt * 16 + 2 * wave) * 32 + lane];
      v16h b1 = Wf[(size_t)(kt * 16 + 2 * wave + 1) * 32 + lane];
      acc00 = wmma_f16(a0, b0, acc00);
      acc01 = wmma_f16(a0, b1, acc01);
      acc10 = wmma_f16(a1, b0, acc10);
      acc11 = wmma_f16(a1, b1, acc11);
      __syncthreads();
    }
  }

  const int nn = lane & 15, half = lane >> 4;
  // helper to fetch accumulator element by (rt, ct, r)
  float accv[2][2][8];
#pragma unroll
  for (int r = 0; r < 8; ++r) {
    accv[0][0][r] = acc00[r]; accv[0][1][r] = acc01[r];
    accv[1][0][r] = acc10[r]; accv[1][1][r] = acc11[r];
  }

  if (EPI == EPI_Q || EPI == EPI_K || EPI == EPI_V) {
#pragma unroll
    for (int rt = 0; rt < 2; ++rt)
#pragma unroll
      for (int ct = 0; ct < 2; ++ct)
#pragma unroll
        for (int r = 0; r < 8; ++r) {
          int m = rt * 16 + half * 8 + r;
          int c = wave * 32 + ct * 16 + nn;
          float v = accv[rt][ct][r] + bias[c];
          if (EPI == EPI_Q)   // Q -> A-panel layout (attention GEMM operand)
            outH[a_panel_idx(rowBase + m, c, 8)] = (_Float16)v;
          else                // K, V -> row-major (kvs + attention epilogue)
            outH[(rowBase + m) * 256 + c] = (_Float16)v;
          if (EPI != EPI_V) sOut[m * 256 + c] = v;
        }
    if (EPI != EPI_V) {     // Frobenius-norm partial + (K only) column sums
      __syncthreads();
      float s2 = 0.f, cs = 0.f;
#pragma unroll
      for (int m = 0; m < 32; ++m) { float v = sOut[m * 256 + tid]; s2 += v * v; cs += v; }
      if (EPI == EPI_K) atomicAdd(&colSum[tid], cs);
      sRed[tid] = s2;
      __syncthreads();
#pragma unroll
      for (int o = 128; o > 0; o >>= 1) {
        if (tid < o) sRed[tid] += sRed[tid + o];
        __syncthreads();
      }
      if (tid == 0) atomicAdd(statSum, sRed[0]);
    }
  } else if (EPI == EPI_LN_RELU) {
#pragma unroll
    for (int rt = 0; rt < 2; ++rt)
#pragma unroll
      for (int ct = 0; ct < 2; ++ct)
#pragma unroll
        for (int r = 0; r < 8; ++r) {
          int m = rt * 16 + half * 8 + r;
          int c = wave * 32 + ct * 16 + nn;
          sOut[m * 256 + c] = accv[rt][ct][r] + bias[c];
        }
    __syncthreads();
    ln_store32(sOut, outF, outH, rowBase, g, bta, wave, lane, true);
  } else {  // EPI_ATTN: (u + N*v)/denom, residual, LN
    if (tid < 32) sRow[tid] = denomB[rowBase + tid];
    __syncthreads();
#pragma unroll
    for (int rt = 0; rt < 2; ++rt)
#pragma unroll
      for (int ct = 0; ct < 2; ++ct)
#pragma unroll
        for (int r = 0; r < 8; ++r) {
          int m = rt * 16 + half * 8 + r;
          int c = wave * 32 + ct * 16 + nn;
          size_t gi = (rowBase + m) * 256 + c;
          float av = (accv[rt][ct][r] + kNf * (float)Vb[gi]) / sRow[m];
          sOut[m * 256 + c] = 0.5f * av + 0.5f * prev[gi];
        }
    __syncthreads();
    ln_store32(sOut, outF, outH, rowBase, g, bta, wave, lane, false);
  }
}

// ---------------------------------------------------------------------------
// Per-row attention normalizer: denom[n] = q[n] . sumKs + N.
// Q is panel-swizzled, so each lane's 16 fragment elements pair with two
// contiguous 8-float runs of sumKs; one shfl_xor(16) combines lane halves.
// ---------------------------------------------------------------------------
__global__ __launch_bounds__(256) void denom_kernel(const _Float16* __restrict__ Qp,
                                                    const float* __restrict__ sumKs,
                                                    float* __restrict__ denom) {
  const int tid = threadIdx.x, lane = tid & 31, wave = tid >> 5;
  const int panel = blockIdx.x * 8 + wave;
  if (panel >= kN / 16) return;
  const int m = lane & 15, half = lane >> 4;
  float acc = 0.f;
  const v16h* Ap = (const v16h*)Qp;
#pragma unroll
  for (int kt = 0; kt < 8; ++kt) {
    v16h q = Ap[(size_t)(panel * 8 + kt) * 32 + lane];
    const float* s0 = sumKs + kt * 32 + half * 8;        // e = 0..7
    const float* s1 = sumKs + kt * 32 + 16 + half * 8;   // e = 8..15
#pragma unroll
    for (int e = 0; e < 8; ++e) acc += (float)q[e] * s0[e];
#pragma unroll
    for (int e = 0; e < 8; ++e) acc += (float)q[8 + e] * s1[e];
  }
  acc += __shfl_xor(acc, 16, 32);
  if (lane < 16) denom[panel * 16 + m] = acc + kNf;
}

// ---------------------------------------------------------------------------
// K^T V split-K: grid.x = 16 output 64x64 tiles, grid.y = 125 chunks of 800
// rows. K-tile stored transposed-into-A-fragment-order at load time, so the
// K^T operand is a contiguous v16h per lane. f32 atomics into KtV[256x256].
// ---------------------------------------------------------------------------
__global__ __launch_bounds__(256) void kvs_kernel(const _Float16* __restrict__ Kb,
                                                  const _Float16* __restrict__ Vb,
                                                  float* __restrict__ KtV) {
  __shared__ __align__(32) _Float16 sKf[4 * 512];  // A^T fragments, m-blocks 0..3
  __shared__ __align__(32) _Float16 sVf[4 * 512];  // B fragments,  n-blocks 0..3
  const int tid = threadIdx.x, lane = tid & 31, wave = tid >> 5;
  const int orow0 = (blockIdx.x >> 2) * 64;   // KtV row block (= K column)
  const int ocol0 = (blockIdx.x & 3) * 64;    // KtV col block (= V column)
  const size_t l0base = (size_t)blockIdx.y * 800;

  const int t0 = wave * 2, t1 = t0 + 1;
  const int mb0 = t0 >> 2, nb0 = t0 & 3;
  const int mb1 = t1 >> 2, nb1 = t1 & 3;
  v8f a0 = {}, a1 = {};

  for (size_t l0 = l0base; l0 < l0base + 800; l0 += 32) {
#pragma unroll
    for (int j = 0; j < 4; ++j) {       // 32 rows x 32 col-pairs, packed loads
      int i = tid + 256 * j;
      int r = i >> 5, cp = i & 31;
      int c = 2 * cp;
      v2h kk = *(const v2h*)(Kb + (l0 + r) * 256 + orow0 + c);
      v2h vv = *(const v2h*)(Vb + (l0 + r) * 256 + ocol0 + c);
      sKf[(c >> 4) * 512 + a_frag_idx(c & 15, r)]             = kk[0];
      sKf[((c + 1) >> 4) * 512 + a_frag_idx((c + 1) & 15, r)] = kk[1];
      sVf[(c >> 4) * 512 + b_frag_idx(c & 15, r)]             = vv[0];
      sVf[((c + 1) >> 4) * 512 + b_frag_idx((c + 1) & 15, r)] = vv[1];
    }
    __syncthreads();
    a0 = wmma_f16(((const v16h*)(sKf + mb0 * 512))[lane],
                  ((const v16h*)(sVf + nb0 * 512))[lane], a0);
    a1 = wmma_f16(((const v16h*)(sKf + mb1 * 512))[lane],
                  ((const v16h*)(sVf + nb1 * 512))[lane], a1);
    __syncthreads();
  }
  const int nn = lane & 15, half = lane >> 4;
#pragma unroll
  for (int r = 0; r < 8; ++r) {
    atomicAdd(&KtV[(orow0 + mb0 * 16 + half * 8 + r) * 256 + ocol0 + nb0 * 16 + nn], a0[r]);
    atomicAdd(&KtV[(orow0 + mb1 * 16 + half * 8 + r) * 256 + ocol0 + nb1 * 16 + nn], a1[r]);
  }
}

// ---------------------------------------------------------------------------
// Output GEMM: h(panel f16) @ W_out(padded 256x64, swizzled) -> out[kN x 40].
// Block: 32 rows x 64 cols, 8 waves -> one 16x16 tile each. No LDS.
// ---------------------------------------------------------------------------
__global__ __launch_bounds__(256) void out_gemm(const _Float16* __restrict__ H,
                                                const _Float16* __restrict__ Wo,
                                                const float* __restrict__ bo,
                                                float* __restrict__ out) {
  const int tid = threadIdx.x, lane = tid & 31, wave = tid >> 5;
  const size_t rowBase = (size_t)blockIdx.x * 32;
  const int rt = wave >> 2, ct = wave & 3;
  const v16h* Ap = (const v16h*)H;
  const v16h* Wf = (const v16h*)Wo;
  const size_t p = (rowBase >> 4) + rt;
  v8f acc = {};
#pragma unroll
  for (int kt = 0; kt < 8; ++kt) {
    acc = wmma_f16(Ap[(p * 8 + kt) * 32 + lane],
                   Wf[(size_t)(kt * 4 + ct) * 32 + lane], acc);
  }
  const int nn = lane & 15, half = lane >> 4;
#pragma unroll
  for (int r = 0; r < 8; ++r) {
    int col = ct * 16 + nn;
    if (col < kClasses)
      out[(rowBase + rt * 16 + half * 8 + r) * kClasses + col] = acc[r] + bo[col];
  }
}

// ------------------------------- utilities ---------------------------------
__global__ void zero_f32(float* __restrict__ p, int n) {
  int i = blockIdx.x * 256 + threadIdx.x;
  if (i < n) p[i] = 0.f;
}
// f32 row-major (KD x 256) -> f16 swizzled [KD/32][16][512]
__global__ void cvt_swz(_Float16* __restrict__ d, const float* __restrict__ s, int KD) {
  int i = blockIdx.x * 256 + threadIdx.x;
  if (i < KD * 256) {
    int k = i >> 8, c = i & 255;
    d[(size_t)((k >> 5) * 16 + (c >> 4)) * 512 + b_frag_idx(c & 15, k & 31)] =
        (_Float16)s[i];
  }
}
// W_out 256x40 -> padded 256x64, swizzled [8][4][512]
__global__ void cvt_pad_out(_Float16* __restrict__ d, const float* __restrict__ s) {
  int i = blockIdx.x * 256 + threadIdx.x;
  if (i < 256 * 64) {
    int k = i >> 6, c = i & 63;
    float v = (c < kClasses) ? s[k * kClasses + c] : 0.f;
    d[(size_t)((k >> 5) * 4 + (c >> 4)) * 512 + b_frag_idx(c & 15, k & 31)] =
        (_Float16)v;
  }
}
// Fold 1/(||q||_F * ||k||_F) into KtV (-> f16 swizzled) and sumK (-> f32)
__global__ void scale_ktv(const float* __restrict__ KtV, _Float16* __restrict__ KtVh,
                          const float* __restrict__ sumK, float* __restrict__ sumKs,
                          const float* __restrict__ sq, const float* __restrict__ sk) {
  int i = blockIdx.x * 256 + threadIdx.x;
  float s = rsqrtf(sq[0]) * rsqrtf(sk[0]);
  if (i < 256 * 256) {
    int k = i >> 8, c = i & 255;
    KtVh[(size_t)((k >> 5) * 16 + (c >> 4)) * 512 + b_frag_idx(c & 15, k & 31)] =
        (_Float16)(KtV[i] * s);
  }
  if (i < 256) sumKs[i] = sumK[i] * s;
}

// ---------------------------------------------------------------------------
extern "C" void kernel_launch(void* const* d_in, const int* in_sizes, int n_in,
                              void* d_out, int out_size, void* d_ws, size_t ws_size,
                              hipStream_t stream) {
  (void)in_sizes; (void)n_in; (void)out_size; (void)ws_size;
  const float* x     = (const float*)d_in[0];
  const float* W_in  = (const float*)d_in[2];
  const float* b_in  = (const float*)d_in[3];
  const float* ln0_g = (const float*)d_in[4];
  const float* ln0_b = (const float*)d_in[5];
  const float* Wq0 = (const float*)d_in[6];  const float* bq0 = (const float*)d_in[7];
  const float* Wk0 = (const float*)d_in[8];  const float* bk0 = (const float*)d_in[9];
  const float* Wv0 = (const float*)d_in[10]; const float* bv0 = (const float*)d_in[11];
  const float* ln1_g = (const float*)d_in[12]; const float* ln1_b = (const float*)d_in[13];
  const float* Wq1 = (const float*)d_in[14]; const float* bq1 = (const float*)d_in[15];
  const float* Wk1 = (const float*)d_in[16]; const float* bk1 = (const float*)d_in[17];
  const float* Wv1 = (const float*)d_in[18]; const float* bv1 = (const float*)d_in[19];
  const float* ln2_g = (const float*)d_in[20]; const float* ln2_b = (const float*)d_in[21];
  const float* W_out = (const float*)d_in[22]; const float* b_out = (const float*)d_in[23];
  float* out = (float*)d_out;

  size_t off = 0;
  auto take = [&](size_t bytes) -> char* {
    char* p = (char*)d_ws + off;
    off += (bytes + 255) & ~(size_t)255;
    return p;
  };
  _Float16* wInH = (_Float16*)take((size_t)kInF * kHid * 2);
  _Float16* wH[6];
  for (int i = 0; i < 6; ++i) wH[i] = (_Float16*)take((size_t)kHid * kHid * 2);
  _Float16* wOutH = (_Float16*)take((size_t)kHid * 64 * 2);
  float*    hA  = (float*)take((size_t)kN * kHid * 4);
  float*    hB  = (float*)take((size_t)kN * kHid * 4);
  _Float16* hAh = (_Float16*)take((size_t)kN * kHid * 2);   // panel-swizzled h
  _Float16* hBh = (_Float16*)take((size_t)kN * kHid * 2);
  _Float16* Qb = (_Float16*)take((size_t)kN * kHid * 2);    // panel-swizzled Q
  _Float16* Kb = (_Float16*)take((size_t)kN * kHid * 2);    // row-major K
  _Float16* Vb = (_Float16*)take((size_t)kN * kHid * 2);    // row-major V
  float* KtV = (float*)take((size_t)(kHid * kHid + kHid + 2) * 4);  // KtV|sumK|sq|sk
  float* sumK = KtV + kHid * kHid;
  float* sq = sumK + kHid;
  float* sk = sq + 1;
  _Float16* KtVh   = (_Float16*)take((size_t)kHid * kHid * 2);
  float*    sumKs  = (float*)take(kHid * 4);
  float*    denomB = (float*)take((size_t)kN * 4);

  // Weight conversion (f32 -> f16, fragment-swizzled)
  cvt_swz<<<(kInF * kHid + 255) / 256, 256, 0, stream>>>(wInH, W_in, kInF);
  const float* wsrc[6] = {Wq0, Wk0, Wv0, Wq1, Wk1, Wv1};
  for (int i = 0; i < 6; ++i)
    cvt_swz<<<(kHid * kHid + 255) / 256, 256, 0, stream>>>(wH[i], wsrc[i], kHid);
  cvt_pad_out<<<(kHid * 64 + 255) / 256, 256, 0, stream>>>(wOutH, W_out);

  // Input layer: hA(+hAh) = relu(LN(x @ W_in + b_in))
  gemm_rows32<EPI_LN_RELU, false, kInF><<<kN / 32, 256, 0, stream>>>(
      x, wInH, b_in, hA, hAh, ln0_g, ln0_b,
      nullptr, nullptr, nullptr, nullptr, nullptr);

  auto run_layer = [&](const float* hIn, const _Float16* hInH,
                       float* hOut, _Float16* hOutH,
                       const _Float16* wq, const float* bq,
                       const _Float16* wk, const float* bk,
                       const _Float16* wv, const float* bv,
                       const float* g, const float* b) {
    zero_f32<<<(kHid * kHid + kHid + 2 + 255) / 256, 256, 0, stream>>>(
        KtV, kHid * kHid + kHid + 2);
    gemm_rows32<EPI_Q, true, kHid><<<kN / 32, 256, 0, stream>>>(
        hInH, wq, bq, nullptr, Qb, nullptr, nullptr, nullptr, nullptr, nullptr, sq, nullptr);
    gemm_rows32<EPI_K, true, kHid><<<kN / 32, 256, 0, stream>>>(
        hInH, wk, bk, nullptr, Kb, nullptr, nullptr, nullptr, nullptr, nullptr, sk, sumK);
    gemm_rows32<EPI_V, true, kHid><<<kN / 32, 256, 0, stream>>>(
        hInH, wv, bv, nullptr, Vb, nullptr, nullptr, nullptr, nullptr, nullptr, nullptr, nullptr);
    kvs_kernel<<<dim3(16, 125), 256, 0, stream>>>(Kb, Vb, KtV);
    scale_ktv<<<(kHid * kHid + 255) / 256, 256, 0, stream>>>(KtV, KtVh, sumK, sumKs, sq, sk);
    denom_kernel<<<(kN / 16 + 7) / 8, 256, 0, stream>>>(Qb, sumKs, denomB);
    gemm_rows32<EPI_ATTN, true, kHid><<<kN / 32, 256, 0, stream>>>(
        Qb, KtVh, nullptr, hOut, hOutH, g, b, Vb, hIn, denomB, nullptr, nullptr);
  };

  run_layer(hA, hAh, hB, hBh, wH[0], bq0, wH[1], bk0, wH[2], bv0, ln1_g, ln1_b);
  run_layer(hB, hBh, hA, hAh, wH[3], bq1, wH[4], bk1, wH[5], bv1, ln2_g, ln2_b);

  // Output: out = h @ W_out + b_out  (A from panel-swizzled f16 copy)
  out_gemm<<<kN / 32, 256, 0, stream>>>(hAh, wOutH, b_out, out);
}
